// DQGN_26834955666046
// MI455X (gfx1250) — matrix operand order repaired and checked by reading, not compiled, not run on hardware
//
#include <hip/hip_runtime.h>
#include <hip/hip_bf16.h>
#include <math.h>

// ---------------------------------------------------------------------------
// GCN forward on gfx1250: bf16 WMMA GEMMs (16x64 tile per wave, A reused
// across 4 N-tiles) + L2-resident coalesced edge scatter with f32 atomics.
// ---------------------------------------------------------------------------

typedef __attribute__((ext_vector_type(16))) __bf16 v16bf;
typedef __attribute__((ext_vector_type(8)))  float  v8f;

#define NNODES  50000
#define NEDGES  800000
#define CIN     128
#define CH      64
#define NGRAPH  256
#define NLAYERS 9   // 1 input layer (CIN->CH) + 8 hidden (CH->CH)

// ---------------- weight prep: f32 [K,Nout] -> bf16 transposed [Nout,K] -----
__global__ void k_wt_convert(const float* __restrict__ w, __bf16* __restrict__ wt,
                             int K, int Nout) {
    int t = blockIdx.x * blockDim.x + threadIdx.x;
    if (t >= K * Nout) return;
    int n = t / K, k = t - n * K;
    wt[t] = (__bf16)w[(size_t)k * Nout + n];
}

// ---------------- degree / normalization -----------------------------------
__global__ void k_fill(float* __restrict__ p, int n, float v) {
    int t = blockIdx.x * blockDim.x + threadIdx.x;
    if (t < n) p[t] = v;
}

__global__ void k_degree(const int* __restrict__ dst, float* __restrict__ deg, int e) {
    int t = blockIdx.x * blockDim.x + threadIdx.x;
    if (t < e) atomicAdd(&deg[dst[t]], 1.0f);
}

__global__ void k_rsqrt(const float* __restrict__ deg, float* __restrict__ dis, int n) {
    int t = blockIdx.x * blockDim.x + threadIdx.x;
    if (t < n) dis[t] = rsqrtf(deg[t]);   // deg >= 1 thanks to self loops
}

// ---------------- WMMA GEMM: Y[rows,64] = X[rows,K] @ W[K,64] ---------------
// Wt is bf16, transposed: Wt[n*K + k].  rows % 16 == 0, K % 32 == 0.
// One wave computes a full 16x64 output tile: A fragment loaded/converted once
// per K-step and reused by 4 WMMAs (one per 16-col N tile).
__global__ void k_gemm_bf16(const float* __restrict__ X, const __bf16* __restrict__ Wt,
                            float* __restrict__ Y, int rows, int K) {
    const int lane = threadIdx.x & 31;
    const int wid  = (blockIdx.x * blockDim.x + threadIdx.x) >> 5;
    if (wid >= (rows >> 4)) return;

    const int m0   = wid << 4;
    const int half = lane >> 4;     // which half-wave
    const int l16  = lane & 15;

    v8f c0 = {}, c1 = {}, c2 = {}, c3 = {};
    const float* xrow = X + (size_t)(m0 + l16) * K;   // A row for this lane

    for (int kk = 0; kk < K; kk += 32) {
        // A fragment (16x32 bf16): lanes 0-15 hold K = kk..kk+15,
        //                          lanes 16-31 hold K = kk+16..kk+31
        const float* xp = xrow + kk + half * 16;
        float4 f0 = *(const float4*)(xp);
        float4 f1 = *(const float4*)(xp + 4);
        float4 f2 = *(const float4*)(xp + 8);
        float4 f3 = *(const float4*)(xp + 12);
        v16bf a;
        a[0]=(__bf16)f0.x; a[1]=(__bf16)f0.y; a[2]=(__bf16)f0.z; a[3]=(__bf16)f0.w;
        a[4]=(__bf16)f1.x; a[5]=(__bf16)f1.y; a[6]=(__bf16)f1.z; a[7]=(__bf16)f1.w;
        a[8]=(__bf16)f2.x; a[9]=(__bf16)f2.y; a[10]=(__bf16)f2.z; a[11]=(__bf16)f2.w;
        a[12]=(__bf16)f3.x; a[13]=(__bf16)f3.y; a[14]=(__bf16)f3.z; a[15]=(__bf16)f3.w;

        // B fragments (32x16 bf16): lane = column, contiguous 16-deep K strip
        const size_t wofs = (size_t)l16 * K + kk + half * 16;
        v16bf b0 = *(const v16bf*)(Wt + wofs);
        v16bf b1 = *(const v16bf*)(Wt + wofs + (size_t)16 * K);
        v16bf b2 = *(const v16bf*)(Wt + wofs + (size_t)32 * K);
        v16bf b3 = *(const v16bf*)(Wt + wofs + (size_t)48 * K);

        c0 = __builtin_amdgcn_wmma_f32_16x16x32_bf16(false, a, false, b0, (short)0, c0, false, false);
        c1 = __builtin_amdgcn_wmma_f32_16x16x32_bf16(false, a, false, b1, (short)0, c1, false, false);
        c2 = __builtin_amdgcn_wmma_f32_16x16x32_bf16(false, a, false, b2, (short)0, c2, false, false);
        c3 = __builtin_amdgcn_wmma_f32_16x16x32_bf16(false, a, false, b3, (short)0, c3, false, false);
    }

    // C layout: VGPR r -> row r (lanes 0-15) / row 8+r (lanes 16-31), N = lane&15
    float* yp = Y + (size_t)m0 * CH;
#pragma unroll
    for (int r = 0; r < 8; ++r) {
        int m = half ? (8 + r) : r;
        float* row = yp + (size_t)m * CH + l16;
        row[0]  = c0[r];
        row[16] = c1[r];
        row[32] = c2[r];
        row[48] = c3[r];
    }
}

// ---------------- self-loop init: acc = dis_i^2 * T -------------------------
__global__ void k_selfloop(const float* __restrict__ T, const float* __restrict__ dis,
                           float* __restrict__ acc, int n) {
    int t = blockIdx.x * blockDim.x + threadIdx.x;
    if (t >= n * CH) return;
    float d = dis[t >> 6];
    acc[t] = d * d * T[t];
}

// ---------------- edge scatter: 64 threads per edge, coalesced --------------
__global__ void k_edge_scatter(const int* __restrict__ src, const int* __restrict__ dst,
                               const float* __restrict__ dis, const float* __restrict__ T,
                               float* __restrict__ acc, int e) {
    int t = blockIdx.x * blockDim.x + threadIdx.x;
    int ei = t >> 6, c = t & 63;
    if (ei >= e) return;
    int s = src[ei], d = dst[ei];
    float w = dis[s] * dis[d];
    atomicAdd(&acc[(size_t)d * CH + c], w * T[(size_t)s * CH + c]);
}

// ---------------- bias + relu ----------------------------------------------
__global__ void k_bias_relu(const float* __restrict__ acc, const float* __restrict__ bias,
                            float* __restrict__ h, int n) {
    int t = blockIdx.x * blockDim.x + threadIdx.x;
    if (t >= n * CH) return;
    float v = acc[t] + bias[t & 63];
    h[t] = v > 0.0f ? v : 0.0f;
}

// ---------------- global add pool ------------------------------------------
__global__ void k_pool(const float* __restrict__ hn, const int* __restrict__ nb,
                       float* __restrict__ pool, int n) {
    int t = blockIdx.x * blockDim.x + threadIdx.x;
    int i = t >> 6, c = t & 63;
    if (i >= n) return;
    atomicAdd(&pool[(size_t)nb[i] * CH + c], hn[t]);
}

// ---------------- readout: tanh(relu(concat(h, rep)) @ w_aggr) --------------
__global__ void k_readout(const float* __restrict__ hn, const float* __restrict__ pt,
                          const int* __restrict__ nb, const float* __restrict__ wa,
                          float* __restrict__ out, int n) {
    int i = blockIdx.x * blockDim.x + threadIdx.x;
    if (i >= n) return;
    const float4* hp = (const float4*)(hn + (size_t)i * CH);
    const float4* pp = (const float4*)(pt + (size_t)nb[i] * CH);
    const float4* wh = (const float4*)(wa);        // first 64 weights for h
    const float4* wr = (const float4*)(wa + CH);   // next 64 for rep
    float z = 0.0f;
#pragma unroll
    for (int j = 0; j < 16; ++j) {
        float4 h4 = hp[j], p4 = pp[j], a4 = wh[j], b4 = wr[j];
        z += fmaxf(h4.x, 0.0f) * a4.x + fmaxf(h4.y, 0.0f) * a4.y +
             fmaxf(h4.z, 0.0f) * a4.z + fmaxf(h4.w, 0.0f) * a4.w;
        z += fmaxf(p4.x, 0.0f) * b4.x + fmaxf(p4.y, 0.0f) * b4.y +
             fmaxf(p4.z, 0.0f) * b4.z + fmaxf(p4.w, 0.0f) * b4.w;
    }
    out[i] = tanhf(z);
}

// ---------------------------------------------------------------------------
extern "C" void kernel_launch(void* const* d_in, const int* in_sizes, int n_in,
                              void* d_out, int out_size, void* d_ws, size_t ws_size,
                              hipStream_t stream) {
    const float* x      = (const float*)d_in[0];   // [N, CIN]
    const int*   eidx   = (const int*)  d_in[1];   // [2, E]
    const int*   nb     = (const int*)  d_in[2];   // [N]
    const float* w0     = (const float*)d_in[3];   // [CIN, CH]
    const float* b0     = (const float*)d_in[4];   // [CH]
    const float* w_hid  = (const float*)d_in[5];   // [NLAYERS-1, CH, CH]
    const float* b_hid  = (const float*)d_in[6];   // [NLAYERS-1, CH]
    const float* w_node = (const float*)d_in[7];   // [CH, CH]
    const float* w_grph = (const float*)d_in[8];   // [CH, CH]
    const float* w_aggr = (const float*)d_in[9];   // [2*CH, 1]
    float* out = (float*)d_out;

    const int* e_src = eidx;
    const int* e_dst = eidx + NEDGES;

    // ---- workspace layout ----
    uintptr_t base = (uintptr_t)d_ws;
    size_t off = 0;
    auto alloc = [&](size_t bytes) -> uintptr_t {
        off = (off + 511) & ~(size_t)511;
        uintptr_t p = base + off;
        off += bytes;
        return p;
    };
    float*  deg   = (float*) alloc((size_t)NNODES * 4);
    float*  dis   = (float*) alloc((size_t)NNODES * 4);
    __bf16* w0t   = (__bf16*)alloc((size_t)CH * CIN * 2);
    __bf16* whidt = (__bf16*)alloc((size_t)(NLAYERS - 1) * CH * CH * 2);
    __bf16* wnt   = (__bf16*)alloc((size_t)CH * CH * 2);
    __bf16* wgt   = (__bf16*)alloc((size_t)CH * CH * 2);
    float*  T     = (float*) alloc((size_t)NNODES * CH * 4);
    float*  acc   = (float*) alloc((size_t)NNODES * CH * 4);
    float*  hA    = (float*) alloc((size_t)NNODES * CH * 4);
    float*  hB    = (float*) alloc((size_t)NNODES * CH * 4);
    float*  pool  = (float*) alloc((size_t)NGRAPH * CH * 4);
    float*  pt    = (float*) alloc((size_t)NGRAPH * CH * 4);
    (void)ws_size;

    const int BT = 256;
    auto nb1 = [&](long n) { return (int)((n + BT - 1) / BT); };

    // ---- weight conversion (transposed bf16) ----
    k_wt_convert<<<nb1(CH * CIN), BT, 0, stream>>>(w0, w0t, CIN, CH);
    for (int i = 0; i < NLAYERS - 1; ++i)
        k_wt_convert<<<nb1(CH * CH), BT, 0, stream>>>(
            w_hid + (size_t)i * CH * CH, whidt + (size_t)i * CH * CH, CH, CH);
    k_wt_convert<<<nb1(CH * CH), BT, 0, stream>>>(w_node, wnt, CH, CH);
    k_wt_convert<<<nb1(CH * CH), BT, 0, stream>>>(w_grph, wgt, CH, CH);

    // ---- degrees (self loop contributes 1 to every node) ----
    k_fill<<<nb1(NNODES), BT, 0, stream>>>(deg, NNODES, 1.0f);
    k_degree<<<nb1(NEDGES), BT, 0, stream>>>(e_dst, deg, NEDGES);
    k_rsqrt<<<nb1(NNODES), BT, 0, stream>>>(deg, dis, NNODES);

    // ---- GCN layers ----
    const long gemm_threads  = (long)(NNODES / 16) * 32;   // one wave per 16x64 tile
    const long gemm_threadsG = (long)(NGRAPH / 16) * 32;
    const long elem_threads  = (long)NNODES * CH;
    const long edge_threads  = (long)NEDGES * CH;

    float* cur = hA;
    float* nxt = hB;
    for (int layer = 0; layer < NLAYERS; ++layer) {
        const float*  in  = (layer == 0) ? x : cur;
        const __bf16* wt  = (layer == 0) ? w0t : (whidt + (size_t)(layer - 1) * CH * CH);
        const float*  bia = (layer == 0) ? b0 : (b_hid + (size_t)(layer - 1) * CH);
        const int     K   = (layer == 0) ? CIN : CH;

        k_gemm_bf16<<<nb1(gemm_threads), BT, 0, stream>>>(in, wt, T, NNODES, K);
        k_selfloop<<<nb1(elem_threads), BT, 0, stream>>>(T, dis, acc, NNODES);
        k_edge_scatter<<<nb1(edge_threads), BT, 0, stream>>>(e_src, e_dst, dis, T, acc, NEDGES);
        k_bias_relu<<<nb1(elem_threads), BT, 0, stream>>>(acc, bia, nxt, NNODES);

        float* tmp = cur; cur = nxt; nxt = tmp;   // layer 0 reads x, so plain
                                                  // ping-pong is always safe
    }

    // ---- node transform (no bias/relu), pool, graph transform ----
    k_gemm_bf16<<<nb1(gemm_threads), BT, 0, stream>>>(cur, wnt, T, NNODES, CH);  // hn = T
    k_fill<<<nb1(NGRAPH * CH), BT, 0, stream>>>(pool, NGRAPH * CH, 0.0f);
    k_pool<<<nb1(elem_threads), BT, 0, stream>>>(T, nb, pool, NNODES);
    k_gemm_bf16<<<nb1(gemm_threadsG), BT, 0, stream>>>(pool, wgt, pt, NGRAPH, CH);

    // ---- readout ----
    k_readout<<<nb1(NNODES), BT, 0, stream>>>(T, pt, nb, w_aggr, out, NNODES);
}
